// EntityEncoder_50826642980970
// MI455X (gfx1250) — matrix operand order repaired and checked by reading, not compiled, not run on hardware
//
#include <hip/hip_runtime.h>

typedef __attribute__((ext_vector_type(16))) _Float16 v16h;
typedef __attribute__((ext_vector_type(8)))  _Float16 v8h;
typedef __attribute__((ext_vector_type(8)))  float    v8f;
typedef __attribute__((ext_vector_type(4)))  float    v4f;

#define DN     512
#define NENT   65536
#define ELEN   39
#define KDIM   192        // 187 dense features padded to 6 K-tiles of 32
#define NMOVES 896
#define KT     6          // K-tiles (192/32)
#define NT     32         // N-tiles (512/16)
#define WDP_BYTES (KDIM * DN * 2)   // 196608 bytes of packed f16 B

// ---------------------------------------------------------------------------
// Prep: pack the 5 stacked dense-weight matrices [192 x 512] into f16 laid out
// in WMMA B-fragment lane order, and fuse the 9 bias vectors into one.
// Flat layout: frag(nt,kt) = 32 lanes x 16 f16, lane-contiguous:
//   idx = ((nt*6 + kt)*32 + lane)*16 + j ;  K = kt*32 + (lane/16)*16 + j ;
//   N = nt*16 + (lane%16)
// ---------------------------------------------------------------------------
__global__ __launch_bounds__(256) void entity_prep_kernel(
    const float* __restrict__ W_level, const float* __restrict__ W_hp,
    const float* __restrict__ W_vol,   const float* __restrict__ W_feats,
    const float* __restrict__ W_bool,
    const float* __restrict__ b_species, const float* __restrict__ b_ability,
    const float* __restrict__ b_item,    const float* __restrict__ b_moveset,
    const float* __restrict__ b_level,   const float* __restrict__ b_hp,
    const float* __restrict__ b_vol,     const float* __restrict__ b_feats,
    const float* __restrict__ b_bool,
    _Float16* __restrict__ WdP, float* __restrict__ bsum)
{
  const int idx = blockIdx.x * 256 + threadIdx.x;
  if (idx < KDIM * DN) {
    const int j    = idx & 15;
    const int lane = (idx >> 4) & 31;
    const int rest = idx >> 9;
    const int kt   = rest % KT;
    const int nt   = rest / KT;
    const int K = kt * 32 + (lane >> 4) * 16 + j;
    const int N = nt * 16 + (lane & 15);
    float v = 0.0f;
    if      (K <   7) v = W_level[K * DN + N];           // level bits (7)
    else if (K <  17) v = W_hp[(K - 7) * DN + N];        // hp bits (10)
    else if (K <  53) v = W_vol[(K - 17) * DN + N];      // volatile bits (36)
    else if (K <  62) v = W_feats[(K - 53) * DN + N];    // scaled feats (9)
    else if (K < 187) v = W_bool[(K - 62) * DN + N];     // one-hots (125)
    WdP[idx] = (_Float16)v;                              // K 187..191 -> 0 pad
  }
  if (idx < DN) {
    bsum[idx] = b_species[idx] + b_ability[idx] + b_item[idx] + b_moveset[idx]
              + b_level[idx] + b_hp[idx] + b_vol[idx] + b_feats[idx] + b_bool[idx];
  }
}

// ---------------------------------------------------------------------------
// Main: 8 waves/block, 16 entities/wave, 128 entities/block, 512 blocks.
// LDS per wave: A tile 16x192 f16 (6144B) + output stage 16x512 f32 (32768B).
// Total dynamic LDS = 8 * 38912 = 311296 B  (<= 320KB per workgroup).
// ---------------------------------------------------------------------------
__global__ __launch_bounds__(256) void entity_encoder_kernel(
    const int*   __restrict__ entity,
    const float* __restrict__ W_species,
    const float* __restrict__ W_ability,
    const float* __restrict__ W_item,
    const float* __restrict__ W_moveset,
    const _Float16* __restrict__ WdP,
    const float* __restrict__ bsum,
    float* __restrict__ out)
{
  extern __shared__ char smem[];
  const int tid  = threadIdx.x;
  const int wave = tid >> 5;
  const int lane = tid & 31;
  const int m    = lane & 15;   // entity-in-tile (row M)
  const int h    = lane >> 4;   // half-wave select

  char* wbase = smem + wave * (6144 + 32768);
  _Float16* A      = (_Float16*)wbase;            // [16][192] f16
  float*    Ostage = (float*)(wbase + 6144);      // [16][512] f32

  // Warm the packed B matrix (192KB, shared by every block) into cache.
  for (int off = tid * 256; off < WDP_BYTES; off += 256 * 256)
    __builtin_prefetch((const char*)WdP + off, 0, 3);

  const int  ebase = blockIdx.x * 128 + wave * 16;
  const int  ent   = ebase + m;
  const int* er    = entity + (size_t)ent * ELEN;
  _Float16*  Ar    = A + m * KDIM;

  // ---- build dense feature row (f16) for this entity; half-wave h builds
  // features [h*96, h*96+96)
  if (h == 0) {
    const int level = er[3], hp = er[4];
#pragma unroll
    for (int i = 0; i < 7;  ++i) Ar[i]     = (_Float16)(float)((level >> i) & 1);
#pragma unroll
    for (int i = 0; i < 10; ++i) Ar[7 + i] = (_Float16)(float)((hp >> i) & 1);
#pragma unroll
    for (int c = 0; c < 9; ++c) {
      const int v = er[30 + c];
#pragma unroll
      for (int b = 0; b < 4; ++b)
        Ar[17 + c * 4 + b] = (_Float16)(float)((v >> b) & 1);
    }
    Ar[53] = (_Float16)((float)level * (1.0f / 100.0f));
    Ar[54] = (_Float16)((float)hp    * (1.0f / 1023.0f));
#pragma unroll
    for (int i = 0; i < 7; ++i) Ar[55 + i] = (_Float16)((float)er[14 + i] * 0.5f);
    const int g = er[5], st = er[6], cb = er[7], tr = er[8], ns = er[9];
    const int tox = er[10], slp = er[11], fn = er[12], ac = er[13];
#pragma unroll
    for (int i = 0; i < 4; ++i) Ar[62 + i] = (_Float16)(float)(g == i);
#pragma unroll
    for (int i = 0; i < 8; ++i) Ar[66 + i] = (_Float16)(float)(st == i);
    Ar[74] = (_Float16)(float)(cb == 0);  Ar[75] = (_Float16)(float)(cb == 1);
    Ar[76] = (_Float16)(float)(tr == 0);  Ar[77] = (_Float16)(float)(tr == 1);
    Ar[78] = (_Float16)(float)(ns == 0);  Ar[79] = (_Float16)(float)(ns == 1);
#pragma unroll
    for (int i = 0; i < 8; ++i) Ar[80 + i] = (_Float16)(float)(tox == i);
#pragma unroll
    for (int i = 0; i < 4; ++i) Ar[88 + i] = (_Float16)(float)(slp == i);
    Ar[92] = (_Float16)(float)(fn == 0);  Ar[93] = (_Float16)(float)(fn == 1);
    Ar[94] = (_Float16)(float)(ac == 0);  Ar[95] = (_Float16)(float)(ac == 1);
  } else {
#pragma unroll
    for (int i = 0; i < 7; ++i) {
      const int b6 = er[14 + i] + 6;          // boost one-hot over 13 bins
#pragma unroll
      for (int j = 0; j < 13; ++j)
        Ar[96 + i * 13 + j] = (_Float16)(float)(b6 == j);
    }
#pragma unroll
    for (int k = 187; k < 192; ++k) Ar[k] = (_Float16)0.0f;   // K pad
  }
  __syncthreads();

  // ---- load the 6 A fragments in contiguous k order (same as packed B):
  // lane l (h=l/16): elems 0..15 -> K = kt*32 + h*16 + j
  v16h af[KT];
#pragma unroll
  for (int kt = 0; kt < KT; ++kt) {
    const v8h lo = *(const v8h*)(A + m * KDIM + kt * 32 + h * 16);
    const v8h hi = *(const v8h*)(A + m * KDIM + kt * 32 + h * 16 + 8);
#pragma unroll
    for (int i = 0; i < 8; ++i) { af[kt][i] = lo[i]; af[kt][8 + i] = hi[i]; }
  }

  // ---- dense GEMM: 32 N-tiles x 6 K-tiles of v_wmma_f32_16x16x32_f16
  const v16h* Bfrag = (const v16h*)WdP;
  for (int nt = 0; nt < NT; ++nt) {
    v8f acc = {};
#pragma unroll
    for (int kt = 0; kt < KT; ++kt) {
      const v16h b = Bfrag[(nt * KT + kt) * 32 + lane];
      acc = __builtin_amdgcn_wmma_f32_16x16x32_f16(
          /*neg_a=*/false, af[kt], /*neg_b=*/false, b,
          /*c_mod=*/(short)0, acc, /*reuse_a=*/false, /*reuse_b=*/false);
    }
    // C layout: lane l, vgpr r -> M = r + 8*h, N = l%16
#pragma unroll
    for (int r = 0; r < 8; ++r)
      Ostage[(r + 8 * h) * DN + nt * 16 + m] = acc[r];
  }
  __syncthreads();

  // ---- writeout: add vocab gathers (L2-hot rows), bias, mask; coalesced f4
  const float* W_mv = W_moveset;
  const float* W_pp = W_moveset + (size_t)NMOVES * DN;
  for (int mm = 0; mm < 16; ++mm) {
    const int  e2  = ebase + mm;
    const int* e2r = entity + (size_t)e2 * ELEN;
    const int sp = e2r[0], ab = e2r[1], it = e2r[2];
    const int mv0 = e2r[21], mv1 = e2r[22], mv2 = e2r[23], mv3 = e2r[24];
    const float q = 1.0f / 1023.0f;
    const float p0 = (float)e2r[25] * q, p1 = (float)e2r[26] * q;
    const float p2 = (float)e2r[27] * q, p3 = (float)e2r[28] * q;
    const bool live = !(sp == 0 || sp == 1);   // SPECIES_NULL / SPECIES_PAD
#pragma unroll
    for (int c = 0; c < 4; ++c) {
      const int col = (lane + c * 32) * 4;     // 128 float4 per row / 32 lanes
      v4f v = *(const v4f*)(Ostage + mm * DN + col);
      v += *(const v4f*)(W_species + (size_t)sp * DN + col);
      v += *(const v4f*)(W_ability + (size_t)ab * DN + col);
      v += *(const v4f*)(W_item    + (size_t)it * DN + col);
      v += *(const v4f*)(W_mv + (size_t)mv0 * DN + col) + p0 * *(const v4f*)(W_pp + (size_t)mv0 * DN + col);
      v += *(const v4f*)(W_mv + (size_t)mv1 * DN + col) + p1 * *(const v4f*)(W_pp + (size_t)mv1 * DN + col);
      v += *(const v4f*)(W_mv + (size_t)mv2 * DN + col) + p2 * *(const v4f*)(W_pp + (size_t)mv2 * DN + col);
      v += *(const v4f*)(W_mv + (size_t)mv3 * DN + col) + p3 * *(const v4f*)(W_pp + (size_t)mv3 * DN + col);
      v += *(const v4f*)(bsum + col);
      if (!live) v = (v4f){0.0f, 0.0f, 0.0f, 0.0f};
      *(v4f*)(out + (size_t)e2 * DN + col) = v;
    }
    if (lane == 0) out[(size_t)NENT * DN + e2] = live ? 1.0f : 0.0f;
  }
}

extern "C" void kernel_launch(void* const* d_in, const int* in_sizes, int n_in,
                              void* d_out, int out_size, void* d_ws, size_t ws_size,
                              hipStream_t stream) {
  (void)in_sizes; (void)n_in; (void)out_size; (void)ws_size;
  const int*   entity    = (const int*)  d_in[0];
  const float* W_species = (const float*)d_in[1];
  const float* b_species = (const float*)d_in[2];
  const float* W_ability = (const float*)d_in[3];
  const float* b_ability = (const float*)d_in[4];
  const float* W_item    = (const float*)d_in[5];
  const float* b_item    = (const float*)d_in[6];
  const float* W_moveset = (const float*)d_in[7];
  const float* b_moveset = (const float*)d_in[8];
  const float* W_level   = (const float*)d_in[9];
  const float* b_level   = (const float*)d_in[10];
  const float* W_hp      = (const float*)d_in[11];
  const float* b_hp      = (const float*)d_in[12];
  const float* W_vol     = (const float*)d_in[13];
  const float* b_vol     = (const float*)d_in[14];
  const float* W_feats   = (const float*)d_in[15];
  const float* b_feats   = (const float*)d_in[16];
  const float* W_bool    = (const float*)d_in[17];
  const float* b_bool    = (const float*)d_in[18];

  _Float16* WdP  = (_Float16*)d_ws;
  float*    bsum = (float*)((char*)d_ws + WDP_BYTES);

  entity_prep_kernel<<<(KDIM * DN + 255) / 256, 256, 0, stream>>>(
      W_level, W_hp, W_vol, W_feats, W_bool,
      b_species, b_ability, b_item, b_moveset, b_level, b_hp, b_vol, b_feats, b_bool,
      WdP, bsum);

  const int blocks = NENT / 128;                     // 512 blocks
  const size_t lds = 8 * (6144 + 32768);             // 311296 B
  entity_encoder_kernel<<<blocks, 256, lds, stream>>>(
      entity, W_species, W_ability, W_item, W_moveset, WdP, bsum, (float*)d_out);
}
